// Net_30855045054764
// MI455X (gfx1250) — matrix operand order repaired and checked by reading, not compiled, not run on hardware
//
#include <hip/hip_runtime.h>

// ---------------- types for WMMA ----------------
typedef __attribute__((ext_vector_type(16))) __bf16 v16bf;
typedef __attribute__((ext_vector_type(8)))  float  v8f;

// float -> bf16 (round-to-nearest-even), stored as raw u16
__device__ __forceinline__ unsigned short f2bf(float f) {
    unsigned int u = __float_as_uint(f);
    u = (u + 0x7FFFu + ((u >> 16) & 1u)) >> 16;
    return (unsigned short)u;
}

// A-fragment layout: identity (contiguous K) so that lanes 0-15 read
// positions [0,16) = K{0..15} and lanes 16-31 read positions [16,32) =
// K{16..31} as ONE contiguous v16bf each, matching the B fragment order.
__device__ __forceinline__ int swz32(int j) {
    return j;
}

// ---------------- constants ----------------
#define BATCH 16
#define C_IN  16
#define V_IN  16000
#define V1    16384
#define V2    4096
#define F1    32
#define F2    64

// ---------------- BatchNorm statistics (per channel over B,V) ----------------
__global__ void bn_stats(const float* __restrict__ x, float* __restrict__ mean,
                         float* __restrict__ rstd) {
    __shared__ float s1[256], s2[256];
    int c = blockIdx.x, t = threadIdx.x;
    float sum = 0.f, sq = 0.f;
    for (int b = 0; b < BATCH; ++b) {
        const float* p = x + ((size_t)b * C_IN + c) * V_IN;
        for (int v = t; v < V_IN; v += 256) { float f = p[v]; sum += f; sq += f * f; }
    }
    s1[t] = sum; s2[t] = sq; __syncthreads();
    for (int o = 128; o > 0; o >>= 1) {
        if (t < o) { s1[t] += s1[t + o]; s2[t] += s2[t + o]; }
        __syncthreads();
    }
    if (t == 0) {
        float n  = (float)BATCH * (float)V_IN;
        float mu = s1[0] / n;
        float va = s2[0] / n - mu * mu;
        mean[c] = mu;
        rstd[c] = rsqrtf(va + 1e-5f);
    }
}

// ---------------- normalize + zero-pad + permute -> X0 (V1, B*C) ----------------
// Also emits bf16 A-fragment slot klocal=0 of the conv1 pair buffer.
__global__ void perm_norm(const float* __restrict__ x, const int* __restrict__ perm,
                          const float* __restrict__ mean, const float* __restrict__ rstd,
                          float* __restrict__ X0, unsigned short* __restrict__ slot) {
    int idx = blockIdx.x * blockDim.x + threadIdx.x;     // V1 * 256
    if (idx >= V1 * 256) return;
    int v = idx >> 8;
    int m = idx & 255;          // m = b*16 + c
    int b = m >> 4, c = m & 15;
    int src = perm[v];
    float r = 0.f;
    if (src < V_IN)
        r = (x[((size_t)b * C_IN + c) * V_IN + src] - mean[c]) * rstd[c];
    X0[(size_t)v * 256 + m] = r;
    slot[((size_t)v * 16 + b) * 32 + swz32(c)] = f2bf(r);  // klocal = 0
}

// ---------------- weight prep: bf16, n-major, B-fragment friendly ----------------
// Wt1[pair p][f][j], j = klocal*16 + c,  value = w1[f, c*16 + (2p+klocal)]
// Wt2[k][f][c],                          value = w2[f, c*16 + k]
__global__ void wprep(const float* __restrict__ w1, const float* __restrict__ w2,
                      unsigned short* __restrict__ Wt1, unsigned short* __restrict__ Wt2) {
    int idx = blockIdx.x * blockDim.x + threadIdx.x;
    if (idx < 8 * 32 * 32) {
        int p = idx >> 10, f = (idx >> 5) & 31, j = idx & 31;
        int klocal = j >> 4, c = j & 15;
        Wt1[idx] = f2bf(w1[f * 256 + c * 16 + (2 * p + klocal)]);
    }
    int i2 = idx - 8 * 32 * 32;
    if (i2 >= 0 && i2 < 16 * 64 * 32) {
        int k = i2 >> 11, f = (i2 >> 5) & 63, c = i2 & 31;
        Wt2[i2] = f2bf(w2[f * 512 + c * 16 + k]);
    }
}

// ---------------- SpMM step of Chebyshev recurrence ----------------
// Xout[v,:] = alpha * (L @ Xin)[v,:] - (useprev ? Xprev[v,:] : 0)
// Also writes bf16 A-fragment copy into pair slot `klocal` (Cc channels/batch).
__global__ void spmm_cheb(const int* __restrict__ cols, const float* __restrict__ vals,
                          const float* __restrict__ Xin, const float* __restrict__ Xprev,
                          float* __restrict__ Xout, unsigned short* __restrict__ slot,
                          int M, int Cc, int klocal, float alpha, int useprev) {
    __shared__ int   scol[16];
    __shared__ float sval[16];
    int v = blockIdx.x, t = threadIdx.x;
    if (t < 16) { scol[t] = cols[v * 16 + t]; sval[t] = vals[v * 16 + t]; }
    __syncthreads();
    for (int m = t; m < M; m += blockDim.x) {
        float acc = 0.f;
        #pragma unroll
        for (int d = 0; d < 16; ++d)
            acc += sval[d] * Xin[(size_t)scol[d] * M + m];
        float r = alpha * acc - (useprev ? Xprev[(size_t)v * M + m] : 0.f);
        Xout[(size_t)v * M + m] = r;
        int b = m / Cc, c = m % Cc;
        slot[((size_t)v * 16 + b) * 32 + swz32(klocal * Cc + c)] = f2bf(r);
    }
}

// ---------------- WMMA GEMM-accumulate: Y(rows,F) += Xb(rows,32) * Wt(F,32)^T ----
// One wave per 16x16 output tile; single v_wmma_f32_16x16x32_bf16 per tile.
__global__ void wmma_gemm(const unsigned short* __restrict__ Xb,
                          const unsigned short* __restrict__ Wt,
                          float* __restrict__ Y,
                          int mtiles, int F, int accumulate) {
    int wave = (blockIdx.x * blockDim.x + threadIdx.x) >> 5;
    int lane = threadIdx.x & 31;
    int ntiles = F >> 4;
    if (wave >= mtiles * ntiles) return;
    int tm = wave / ntiles;
    int tn = wave - tm * ntiles;
    int hi = lane >> 4;                         // 0: lanes 0-15, 1: lanes 16-31
    int m  = tm * 16 + (lane & 15);
    int n  = tn * 16 + (lane & 15);

    v16bf a = *(const v16bf*)(Xb + (size_t)m * 32 + hi * 16);  // pre-swizzled A
    v16bf b = *(const v16bf*)(Wt + (size_t)n * 32 + hi * 16);  // contiguous K run

    int row0 = tm * 16 + hi * 8;
    size_t base = (size_t)row0 * F + n;
    v8f c;
    if (accumulate) {
        #pragma unroll
        for (int g = 0; g < 8; ++g) c[g] = Y[base + (size_t)g * F];
    } else {
        #pragma unroll
        for (int g = 0; g < 8; ++g) c[g] = 0.f;
    }
    v8f d = __builtin_amdgcn_wmma_f32_16x16x32_bf16(false, a, false, b,
                                                    (short)0, c, false, false);
    #pragma unroll
    for (int g = 0; g < 8; ++g) Y[base + (size_t)g * F] = d[g];
}

// ---------------- bias + ReLU + pool4 after conv1 -> conv2 input ----------------
__global__ void pool_relu1(const float* __restrict__ Y1, const float* __restrict__ bias,
                           float* __restrict__ P, unsigned short* __restrict__ slot) {
    int idx = blockIdx.x * blockDim.x + threadIdx.x;     // V2 * 16 * 32
    if (idx >= V2 * 512) return;
    int v2 = idx >> 9;
    int m  = idx & 511;          // b*32 + f
    int b = m >> 5, f = m & 31;
    float mx = -3.402823466e38f;
    #pragma unroll
    for (int q = 0; q < 4; ++q)
        mx = fmaxf(mx, Y1[(((size_t)(v2 * 4 + q)) * 16 + b) * F1 + f]);
    float r = fmaxf(mx + bias[f], 0.f);
    P[(size_t)v2 * 512 + m] = r;
    slot[((size_t)v2 * 16 + b) * 32 + swz32(f)] = f2bf(r);  // conv2 k=0 slot
}

// ---------------- bias + ReLU + pool4 after conv2 -> fc input (b, f, v) ---------
__global__ void pool_relu2(const float* __restrict__ Y2, const float* __restrict__ bias,
                           float* __restrict__ Xfc) {
    int idx = blockIdx.x * blockDim.x + threadIdx.x;     // 16 * 64 * 1024
    if (idx >= BATCH * F2 * 1024) return;
    int b = idx >> 16;
    int f = (idx >> 10) & 63;
    int v = idx & 1023;
    float mx = -3.402823466e38f;
    #pragma unroll
    for (int q = 0; q < 4; ++q)
        mx = fmaxf(mx, Y2[(((size_t)(v * 4 + q)) * 16 + b) * F2 + f]);
    Xfc[idx] = fmaxf(mx + bias[f], 0.f);
}

// ---------------- FC1: out(16,512) = Xfc(16,65536) @ W^T + b ----------------
__global__ void fc_kernel(const float* __restrict__ Xfc, const float* __restrict__ W,
                          const float* __restrict__ bias, float* __restrict__ out) {
    __shared__ float red[16][257];
    int o = blockIdx.x, t = threadIdx.x;
    float acc[16];
    #pragma unroll
    for (int b = 0; b < 16; ++b) acc[b] = 0.f;
    const float* wr = W + (size_t)o * 65536;
    for (int i = t; i < 65536; i += 256) {
        float w = wr[i];
        #pragma unroll
        for (int b = 0; b < 16; ++b) acc[b] += w * Xfc[(size_t)b * 65536 + i];
    }
    #pragma unroll
    for (int b = 0; b < 16; ++b) red[b][t] = acc[b];
    __syncthreads();
    if (t < 16) {
        float s = 0.f;
        for (int i = 0; i < 256; ++i) s += red[t][i];
        out[(size_t)t * 512 + o] = s + bias[o];
    }
}

// ---------------- orchestration ----------------
extern "C" void kernel_launch(void* const* d_in, const int* in_sizes, int n_in,
                              void* d_out, int out_size, void* d_ws, size_t ws_size,
                              hipStream_t stream) {
    const float* x    = (const float*)d_in[0];
    const int*   perm = (const int*)  d_in[1];
    const int*   l1c  = (const int*)  d_in[3];
    const float* l1v  = (const float*)d_in[4];
    const int*   l2c  = (const int*)  d_in[6];
    const float* l2v  = (const float*)d_in[7];
    const float* w1   = (const float*)d_in[8];
    const float* b1   = (const float*)d_in[9];
    const float* w2   = (const float*)d_in[10];
    const float* b2   = (const float*)d_in[11];
    const float* fcw  = (const float*)d_in[12];
    const float* fcb  = (const float*)d_in[13];

    char* p = (char*)d_ws;
    auto carve = [&](size_t bytes) -> void* {
        void* r = (void*)p;
        p += (bytes + 255) & ~(size_t)255;
        return r;
    };
    float*          mean  = (float*)carve(16 * 4);
    float*          rstd  = (float*)carve(16 * 4);
    unsigned short* Wt1   = (unsigned short*)carve((size_t)8 * 32 * 32 * 2);
    unsigned short* Wt2   = (unsigned short*)carve((size_t)16 * 64 * 32 * 2);
    float*          XA    = (float*)carve((size_t)V1 * 256 * 4);
    float*          XB    = (float*)carve((size_t)V1 * 256 * 4);
    float*          XC    = (float*)carve((size_t)V1 * 256 * 4);
    unsigned short* pair1 = (unsigned short*)carve((size_t)V1 * 16 * 32 * 2); // 16.8 MB
    float*          Y     = (float*)carve((size_t)V1 * 16 * F1 * 4);          // 33.6 MB
    // aliases (stream-serialized lifetimes)
    unsigned short* slot2 = pair1;                              // conv2 A slab (4.2 MB)
    float*          Xfc   = (float*)((char*)pair1 + 8388608);   // 2nd half of pair1 region
    float* Xc[3] = {XA, XB, XC};

    bn_stats <<<16,    256, 0, stream>>>(x, mean, rstd);
    perm_norm<<<16384, 256, 0, stream>>>(x, perm, mean, rstd, Xc[0], pair1);
    wprep    <<<160,   256, 0, stream>>>(w1, w2, Wt1, Wt2);

    // conv1: Chebyshev recurrence, WMMA-accumulate per order pair (K reduction = 32)
    for (int k = 1; k < 16; ++k) {
        spmm_cheb<<<V1, 256, 0, stream>>>(l1c, l1v, Xc[(k - 1) % 3], Xc[(k + 1) % 3],
                                          Xc[k % 3], pair1, 256, 16, k & 1,
                                          (k == 1) ? 1.f : 2.f, k >= 2);
        if (k & 1) {
            int pidx = k >> 1;   // pair (k-1, k)
            wmma_gemm<<<4096, 256, 0, stream>>>(pair1, Wt1 + pidx * 1024, Y,
                                                V1 * 16 / 16, F1, pidx > 0);
        }
    }
    pool_relu1<<<8192, 256, 0, stream>>>(Y, b1, Xc[0], slot2);

    // conv2: 32 in-channels -> one WMMA K-slab per Chebyshev order
    wmma_gemm<<<2048, 256, 0, stream>>>(slot2, Wt2, Y, V2 * 16 / 16, F2, 0);
    for (int k = 1; k < 16; ++k) {
        spmm_cheb<<<V2, 256, 0, stream>>>(l2c, l2v, Xc[(k - 1) % 3], Xc[(k + 1) % 3],
                                          Xc[k % 3], slot2, 512, 32, 0,
                                          (k == 1) ? 1.f : 2.f, k >= 2);
        wmma_gemm<<<2048, 256, 0, stream>>>(slot2, Wt2 + k * 2048, Y,
                                            V2 * 16 / 16, F2, 1);
    }
    pool_relu2<<<4096, 256, 0, stream>>>(Y, b2, Xfc);
    fc_kernel <<<512,  256, 0, stream>>>(Xfc, fcw, fcb, (float*)d_out);
}
